// DynamicTimeWarpingLoss_59296318488925
// MI455X (gfx1250) — matrix operand order repaired and checked by reading, not compiled, not run on hardware
//
#include <hip/hip_runtime.h>
#include <hip/hip_bf16.h>
#include <math.h>

#define NB 64
#define NT 384
#define ND 512
#define JT 4          // tiles along j per wave (16x64 output strip)

typedef __attribute__((ext_vector_type(16))) __bf16 v16bf;
typedef __attribute__((ext_vector_type(8)))  float  v8f;

// ---------------------------------------------------------------------------
// Phase 1: pairwise L2 cost. One wave32 computes a 16x64 strip: the A fragment
// (pred rows, converted f32->bf16) is built once per K-step and reused by 4
// v_wmma_f32_16x16x32_bf16 issues. Row norms are accumulated exactly in f32
// alongside the conversions, so only the cross term -2*p.q is bf16-rounded
// (~1e-5 relative error on the resulting distance).
// ---------------------------------------------------------------------------
__global__ __launch_bounds__(32)
void dtw_cost_wmma_kernel(const float* __restrict__ pred,
                          const float* __restrict__ targ,
                          float* __restrict__ cost) {
  const int i0   = blockIdx.x * 16;
  const int j0   = blockIdx.y * (16 * JT);
  const int b    = blockIdx.z;
  const int lane = threadIdx.x;   // 0..31 (wave32)
  const int half = lane >> 4;     // 0 or 1
  const int mr   = lane & 15;

  const float* __restrict__ pa =
      pred + ((size_t)b * NT + i0 + mr) * ND;            // A row (M = mr)
  const float* __restrict__ pb[JT];
  #pragma unroll
  for (int t = 0; t < JT; ++t)
    pb[t] = targ + ((size_t)b * NT + j0 + t * 16 + mr) * ND;  // B col (N = mr)

  v8f acc[JT];
  #pragma unroll
  for (int t = 0; t < JT; ++t) acc[t] = (v8f){};
  float sp = 0.0f;       // partial ||pred row||^2 (this half's K slices)
  float sq[JT] = {};     // partial ||targ row||^2 per j-tile

  for (int k0 = 0; k0 < ND; k0 += 32) {
    // A 16x32 bf16 layout: elems 0..7 -> K = 8*half+e ; elems 8..15 -> K = 16+8*half+e
    v16bf afrag;
    #pragma unroll
    for (int e = 0; e < 8; ++e) {
      float x0 = pa[k0 + half * 8 + e];
      float x1 = pa[k0 + 16 + half * 8 + e];
      afrag[e]     = (__bf16)x0;
      afrag[8 + e] = (__bf16)x1;
      sp += x0 * x0 + x1 * x1;
    }
    // B 32x16 bf16 layout: elems 0..7 -> K = 8*half+e ; elems 8..15 -> K = 16+8*half+(e-8)
    #pragma unroll
    for (int t = 0; t < JT; ++t) {
      v16bf bfrag;
      #pragma unroll
      for (int e = 0; e < 16; ++e) {
        float y = pb[t][k0 + (e & 8) * 2 + half * 8 + (e & 7)];
        bfrag[e] = (__bf16)y;
        sq[t] += y * y;
      }
      acc[t] = __builtin_amdgcn_wmma_f32_16x16x32_bf16(
          /*neg_a=*/false, afrag, /*neg_b=*/false, bfrag,
          /*c_mod=*/(short)0, acc[t], /*reuse_a=*/false, /*reuse_b=*/false);
    }
  }

  // Lanes L and L^16 hold complementary K-halves of the same row -> combine.
  sp += __shfl_xor(sp, 16, 32);                  // ||pred[i0+mr]||^2
  #pragma unroll
  for (int t = 0; t < JT; ++t)
    sq[t] += __shfl_xor(sq[t], 16, 32);          // ||targ[j0+t*16+mr]||^2

  #pragma unroll
  for (int t = 0; t < JT; ++t) {
    float* __restrict__ out = cost + ((size_t)b * NT + i0) * NT + j0 + t * 16;
    #pragma unroll
    for (int r = 0; r < 8; ++r) {
      const int M  = r + 8 * half;               // C/D layout: VGPR r -> M = r+8*half
      float npM    = __shfl(sp, M, 32);          // ||pred row M||^2 from lane M
      float sqd    = npM + sq[t] - 2.0f * acc[t][r];
      out[(size_t)M * NT + mr] = sqrtf(fmaxf(sqd, 0.0f));
    }
  }
}

// ---------------------------------------------------------------------------
// Phase 2: DTW min-plus DP, anti-diagonal wavefront. One workgroup per batch,
// 384 threads (12 waves), rotating 3-diagonal buffers in LDS, one barrier per
// anti-diagonal (767 total, latency-bound, 64 batches in parallel).
// ---------------------------------------------------------------------------
__global__ __launch_bounds__(NT)
void dtw_dp_kernel(const float* __restrict__ cost, float* __restrict__ dists) {
  __shared__ float diag[3][NT];
  const int b = blockIdx.x;
  const int i = threadIdx.x;
  const float INF = __builtin_inff();
  const float* __restrict__ C = cost + (size_t)b * NT * NT;

  for (int d = 0; d < 2 * NT - 1; ++d) {
    float*       cur = diag[d % 3];
    const float* p1  = diag[(d + 2) % 3];   // diagonal d-1
    const float* p2  = diag[(d + 1) % 3];   // diagonal d-2
    const int j = d - i;
    if (j >= 0 && j < NT) {
      float c = C[(size_t)i * NT + j];
      float v;
      if (d == 0) {
        v = c;                                   // dp[0][0] = cost[0][0]
      } else {
        float up = (i > 0)          ? p1[i - 1] : INF;   // dp[i-1][j]
        float lf = (j > 0)          ? p1[i]     : INF;   // dp[i][j-1]
        float ul = (i > 0 && j > 0) ? p2[i - 1] : INF;   // dp[i-1][j-1]
        v = c + fminf(fminf(up, lf), ul);
      }
      cur[i] = v;
    }
    __syncthreads();
  }
  if (i == 0) dists[b] = diag[(2 * NT - 2) % 3][NT - 1];
}

// ---------------------------------------------------------------------------
// Phase 3: exact mean over 64 batch distances (deterministic, no atomics).
// ---------------------------------------------------------------------------
__global__ __launch_bounds__(64)
void dtw_mean_kernel(const float* __restrict__ dists, float* __restrict__ out) {
  if (threadIdx.x == 0) {
    float s = 0.0f;
    for (int b = 0; b < NB; ++b) s += dists[b];
    out[0] = s / (float)NB;
  }
}

extern "C" void kernel_launch(void* const* d_in, const int* in_sizes, int n_in,
                              void* d_out, int out_size, void* d_ws, size_t ws_size,
                              hipStream_t stream) {
  (void)in_sizes; (void)n_in; (void)out_size;
  const float* pred = (const float*)d_in[0];
  const float* targ = (const float*)d_in[1];
  float* out  = (float*)d_out;

  const size_t cost_elems = (size_t)NB * NT * NT;       // ~36 MB of f32
  if (ws_size < (cost_elems + NB) * sizeof(float)) return;
  float* cost  = (float*)d_ws;
  float* dists = cost + cost_elems;

  dim3 gcost(NT / 16, NT / (16 * JT), NB);
  dtw_cost_wmma_kernel<<<gcost, 32, 0, stream>>>(pred, targ, cost);
  dtw_dp_kernel<<<NB, NT, 0, stream>>>(cost, dists);
  dtw_mean_kernel<<<1, 64, 0, stream>>>(dists, out);
}
